// RLIFLayer_10617159156208
// MI455X (gfx1250) — matrix operand order repaired and checked by reading, not compiled, not run on hardware
//
#include <hip/hip_runtime.h>

typedef __attribute__((ext_vector_type(16))) __bf16    bf16x16;
typedef __attribute__((ext_vector_type(8)))  __bf16    bf16x8;
typedef __attribute__((ext_vector_type(8)))  float     fx8;
typedef __attribute__((ext_vector_type(4)))  unsigned  u32x4;
typedef __attribute__((ext_vector_type(8)))  int       i32x8;
typedef __attribute__((ext_vector_type(4)))  int       i32x4;

#define ALPHA_MIN 0.8187307530779818f
#define ALPHA_MAX 0.9607894391523232f

#if defined(__has_builtin)
#if __has_builtin(__builtin_amdgcn_tensor_load_to_lds) && \
    __has_builtin(__builtin_amdgcn_s_wait_tensorcnt)
#define USE_TDM 1
#endif
#endif
#ifndef USE_TDM
#define USE_TDM 0
#endif

// float -> bf16 round-to-nearest-even (only used in one-time pack kernels)
__device__ __forceinline__ __bf16 f2bf(float f) {
    unsigned u = __builtin_bit_cast(unsigned, f);
    unsigned r = u + 0x7FFFu + ((u >> 16) & 1u);
    unsigned short h = (unsigned short)(r >> 16);
    return __builtin_bit_cast(__bf16, h);
}

// ---------------------------------------------------------------------------
// Fragment layouts (CDNA5 ISA §7.12.2), bf16 16x16x32:
//  A (16x32 MxK): row m = lane%16; elem e -> K = (lane/16)*16 + e (order chosen here, same as B)
//  B (32x16 KxN): col n = lane%16; elem e -> K = (lane/16)*16 + e (order chosen here, same as A)
//  C/D (16x16):   elem r -> M = r + 8*(lane/16), N = lane%16
// Fragment storage: [tileIdx][lane(32)][e(16)] bf16, 32 B per lane.
// ---------------------------------------------------------------------------

// Phase 0a: pack x[32000,256] into A fragments [mtile=2000][kt=8][lane][e]
__global__ __launch_bounds__(256) void x_pack_kernel(
    const float* __restrict__ x, __bf16* __restrict__ xf) {
    int idx   = blockIdx.x * 256 + threadIdx.x;   // 0..511999 lane-fragments
    int lane  = idx & 31;
    int kt    = (idx >> 5) & 7;
    int mtile = idx >> 8;
    int half  = lane >> 4, nn = lane & 15;
    const float* row = x + (size_t)(mtile * 16 + nn) * 256 + kt * 32;
    const float* p0  = row + half * 16;
    const float* p1  = row + half * 16 + 8;
    bf16x16 a;
#pragma unroll
    for (int e = 0; e < 8; ++e) { a[e] = f2bf(p0[e]); a[e + 8] = f2bf(p1[e]); }
    ((bf16x16*)xf)[idx] = a;
}

// Phase 0b: pack W[512,256] into B fragments [ntile=32][kt=8][lane][e]
__global__ __launch_bounds__(256) void w_pack_kernel(
    const float* __restrict__ W, __bf16* __restrict__ wf) {
    int idx   = blockIdx.x * 256 + threadIdx.x;   // 0..8191
    int lane  = idx & 31;
    int kt    = (idx >> 5) & 7;
    int ntile = idx >> 8;
    int half  = lane >> 4, nn = lane & 15;
    const float* p = W + (size_t)(ntile * 16 + nn) * 256 + kt * 32 + half * 16;
    bf16x16 b;
#pragma unroll
    for (int e = 0; e < 16; ++e) b[e] = f2bf(p[e]);
    ((bf16x16*)wf)[idx] = b;
}

// Phase 0c: pack Vz (V, zero diag) into B fragments [ntile=32][kt=16][lane][e]
__global__ __launch_bounds__(256) void vz_pack_kernel(
    const float* __restrict__ V, __bf16* __restrict__ vf) {
    int idx   = blockIdx.x * 256 + threadIdx.x;   // 0..16383
    int lane  = idx & 31;
    int kt    = (idx >> 5) & 15;
    int ntile = idx >> 9;
    int half  = lane >> 4, nn = lane & 15;
    int gn = ntile * 16 + nn;
    bf16x16 b;
#pragma unroll
    for (int e = 0; e < 16; ++e) {
        int gk  = kt * 32 + half * 16 + e;
        float v = (gk == gn) ? 0.0f : V[(size_t)gk * 512 + gn];
        b[e] = f2bf(v);
    }
    ((bf16x16*)vf)[idx] = b;
}

// ---------------------------------------------------------------------------
// Phase 1: Wx[t][b][h] GEMM from pre-packed fragments. Pure load+wmma loop.
// ---------------------------------------------------------------------------
__global__ __launch_bounds__(256) void wx_gemm_kernel(
    const bf16x16* __restrict__ xf, const bf16x16* __restrict__ wf,
    float* __restrict__ Wx) {
    const int lane  = threadIdx.x & 31;
    const int wave  = threadIdx.x >> 5;
    const int nn    = lane & 15;
    const int half  = lane >> 4;
    const int mtile = blockIdx.x >> 2;                  // 0..1999
    const int ntile = ((blockIdx.x & 3) << 3) + wave;   // 0..31

    fx8 acc = {};
#pragma unroll
    for (int kt = 0; kt < 8; ++kt) {
        bf16x16 a = xf[((size_t)mtile * 8 + kt) * 32 + lane];
        bf16x16 b = wf[((size_t)ntile * 8 + kt) * 32 + lane];
        acc = __builtin_amdgcn_wmma_f32_16x16x32_bf16(
            false, a, false, b, (short)0, acc, false, false);
    }
#pragma unroll
    for (int r = 0; r < 8; ++r) {
        unsigned row = (unsigned)(mtile * 16 + r + half * 8);
        unsigned bb  = row / 1000u;
        unsigned tt  = row - bb * 1000u;
        Wx[((size_t)tt * 32 + bb) * 512 + ntile * 16 + nn] = acc[r];
    }
}

#if USE_TDM
// Issue TDM: contiguous nelem f32 from gptr -> LDS offset lds_off.
// D# per CDNA5 ISA: count=1, type=2(image),
// data_size=2(4B), tile_dim0=nelem, tile/tensor_dim1=1, stride=nelem.
// 6-arg builtin form (clang-23 / therock-10.0): groups 2/3 + extra group
// zero-filled (2D tensor -> higher-dim groups disabled).
__device__ __forceinline__ void tdm_load_f32(unsigned lds_off,
                                             const void* gptr,
                                             unsigned nelem) {
    unsigned long long ga = (unsigned long long)(size_t)gptr;
    u32x4 g0;
    g0[0] = 0x1u;                                          // count=1
    g0[1] = lds_off;                                       // lds_addr
    g0[2] = (unsigned)(ga & 0xFFFFFFFFu);                  // global_addr lo
    g0[3] = (unsigned)((ga >> 32) & 0x01FFFFFFu) | (2u << 30);  // hi | type=2
    i32x8 g1;
    g1[0] = (int)(2u << 16);                               // data_size=4B
    g1[1] = (int)((nelem & 0xFFFFu) << 16);                // tensor_dim0 lo16
    g1[2] = (int)((nelem >> 16) & 0xFFFFu) | (1 << 16);    // dim0 hi | dim1=1
    g1[3] = (int)((nelem & 0xFFFFu) << 16);                // tile_dim0
    g1[4] = 1;                                             // tile_dim1=1
    g1[5] = (int)nelem;                                    // dim0_stride lo32
    g1[6] = 0;
    g1[7] = 0;
    i32x4 z4 = {0, 0, 0, 0};
    i32x8 z8 = {0, 0, 0, 0, 0, 0, 0, 0};
    __builtin_amdgcn_tensor_load_to_lds(g0, g1, z4, z4, z8, 0);
}
#endif

// ---------------------------------------------------------------------------
// Phase 2: persistent recurrent scan. 2 workgroups x 1024 threads.
// Spikes double-buffered in LDS; Wx slab (32KB/step) DMA'd by TDM into an
// LDS double-buffer, overlapped with the 16 WMMAs of the same step.
// ---------------------------------------------------------------------------
__global__ __launch_bounds__(1024) void rlif_scan_kernel(
    const float* __restrict__ Wx,            // [1000][32][512]
    const bf16x16* __restrict__ vf,          // packed Vz fragments
    const float* __restrict__ alpha,
    const float* __restrict__ v_thresh,
    float* __restrict__ out) {               // [32][1000][512]
    __shared__ __align__(32) __bf16 s_lds[2][16][512];     // 32 KB spikes
    __shared__ __align__(16) float  wx_stage[2][16 * 512]; // 64 KB Wx stage

    const int lane = threadIdx.x & 31;
    const int wave = threadIdx.x >> 5;    // ntile 0..31
    const int nn   = lane & 15;
    const int half = lane >> 4;
    const int wg   = blockIdx.x;          // 0..1
    const int h    = wave * 16 + nn;

    float a = alpha[h];
    a = fminf(fmaxf(a, ALPHA_MIN), ALPHA_MAX);
    const float oma = 1.0f - a;
    const float thr = v_thresh[h];

    // s_{-1} = 0
    for (int i = threadIdx.x; i < 16 * 512; i += 1024)
        ((unsigned short*)s_lds[0])[i] = 0;

#if USE_TDM
    if (wave == 0) {   // preload t=0 slab
        tdm_load_f32((unsigned)(size_t)(void*)&wx_stage[0][0],
                     Wx + (size_t)wg * 16 * 512, 16 * 512);
        __builtin_amdgcn_s_wait_tensorcnt(0);
    }
#endif
    __syncthreads();

    fx8 u = {};   // membrane potential, elem r -> (m = r+8*half, n = nn)
    fx8 s = {};   // previous spike, same mapping

    for (int t = 0; t < 1000; ++t) {
        const int cur = t & 1;
        const int nxt = cur ^ 1;

#if USE_TDM
        // kick off DMA of next step's Wx slab; overlaps with WMMAs below
        if (wave == 0 && t + 1 < 1000)
            tdm_load_f32((unsigned)(size_t)(void*)&wx_stage[nxt][0],
                         Wx + ((size_t)(t + 1) * 32 + wg * 16) * 512, 16 * 512);
#endif
        // rec = S @ Vz for this wave's 16 columns
        fx8 acc = {};
#pragma unroll 4
        for (int kt = 0; kt < 16; ++kt) {
            const int k0 = kt * 32;
            bf16x8 lo = *(const bf16x8*)&s_lds[cur][nn][k0 + half * 16];
            bf16x8 hi = *(const bf16x8*)&s_lds[cur][nn][k0 + half * 16 + 8];
            bf16x16 afrag = __builtin_shufflevector(
                lo, hi, 0, 1, 2, 3, 4, 5, 6, 7, 8, 9, 10, 11, 12, 13, 14, 15);
            bf16x16 bfrag = vf[(wave * 16 + kt) * 32 + lane];
            acc = __builtin_amdgcn_wmma_f32_16x16x32_bf16(
                false, afrag, false, bfrag, (short)0, acc, false, false);
        }

#if !USE_TDM
        const float* wxt = Wx + ((size_t)t * 32 + wg * 16) * 512;
#endif
#pragma unroll
        for (int r = 0; r < 8; ++r) {
            const int m = r + half * 8;
#if USE_TDM
            float wx = wx_stage[cur][m * 512 + h];
#else
            float wx = wxt[m * 512 + h];
#endif
            float un = a * (u[r] - s[r]) + oma * (wx + acc[r]);
            u[r] = un;
            bool fire = un > thr;
            float sp  = fire ? 1.0f : 0.0f;
            s[r] = sp;
            out[((size_t)(wg * 16 + m) * 1000 + t) * 512 + h] = sp;
            // bf16 1.0 = 0x3F80; avoid cvt chain
            ((unsigned short*)s_lds[nxt])[m * 512 + h] =
                fire ? (unsigned short)0x3F80 : (unsigned short)0;
        }
#if USE_TDM
        if (wave == 0) __builtin_amdgcn_s_wait_tensorcnt(0);
#endif
        __syncthreads();
    }
}

// ---------------------------------------------------------------------------
extern "C" void kernel_launch(void* const* d_in, const int* in_sizes, int n_in,
                              void* d_out, int out_size, void* d_ws,
                              size_t ws_size, hipStream_t stream) {
    (void)in_sizes; (void)n_in; (void)out_size; (void)ws_size;
    const float* x        = (const float*)d_in[0];
    const float* W        = (const float*)d_in[1];
    const float* V        = (const float*)d_in[2];
    const float* alpha    = (const float*)d_in[3];
    const float* v_thresh = (const float*)d_in[4];
    float* out = (float*)d_out;

    // Workspace: [0,512K) Vz frags | [512K,768K) W frags |
    //            [1M,1M+15.7M) x frags | [17M, 17M+65.6M) Wx f32
    char* ws = (char*)d_ws;
    __bf16* vfrag = (__bf16*)(ws);
    __bf16* wfrag = (__bf16*)(ws + 0x80000);
    __bf16* xfrag = (__bf16*)(ws + 0x100000);
    float*  Wx    = (float*)(ws + 0x1100000);

    vz_pack_kernel<<<64,   256, 0, stream>>>(V, vfrag);
    w_pack_kernel <<<32,   256, 0, stream>>>(W, wfrag);
    x_pack_kernel <<<2000, 256, 0, stream>>>(x, xfrag);
    wx_gemm_kernel<<<8000, 256, 0, stream>>>(
        (const bf16x16*)xfrag, (const bf16x16*)wfrag, Wx);
    rlif_scan_kernel<<<2, 1024, 0, stream>>>(
        Wx, (const bf16x16*)vfrag, alpha, v_thresh, out);
}
